// Attend_54279796687404
// MI455X (gfx1250) — matrix operand order repaired and checked by reading, not compiled, not run on hardware
//
#include <hip/hip_runtime.h>
#include <hip/hip_bf16.h>

typedef _Float16 v16h __attribute__((ext_vector_type(16)));
typedef float    v8f  __attribute__((ext_vector_type(8)));
typedef unsigned v4u  __attribute__((ext_vector_type(4)));
typedef int      v8i  __attribute__((ext_vector_type(8)));

#define N_SEQ   2048
#define DHEAD   64
#define BLOCK_M 128
#define BLOCK_N 64
#define NWAVES  8
#define LOG2E   1.4426950408889634f
#define SC      0.123091490979332845f   /* (64+2)^-0.5 */
#define NEGBIG  (-3.0e38f)
#define TILE_ELEMS (BLOCK_N * DHEAD)    /* 4096 f32 per K or V tile */

__device__ __forceinline__ unsigned pack2h(_Float16 a, _Float16 b) {
    union { _Float16 h[2]; unsigned u; } x;
    x.h[0] = a; x.h[1] = b;
    return x.u;
}

// Tensor Data Mover: 1-D DMA of `nelem` f32 from global to LDS.
// D# per CDNA5 ISA: group0 = {flags, lds_addr, global_addr, type=2},
// group1 = {data_size=4B, tensor_dim0=nelem, tensor_dim1=1, tile_dim0=nelem, strides}.
// 2-group form (VADDR2/3 = NULL) since the tensor is <= 2D. Tracked on TENSORcnt.
__device__ __forceinline__ void tdm_load_1d(const float* gptr, unsigned lds_off,
                                            unsigned nelem) {
    unsigned long long ga = (unsigned long long)(const void*)gptr;
    v4u g0;
    g0[0] = 1u;                                   // count=1, user descriptor
    g0[1] = lds_off;                              // LDS byte address
    g0[2] = (unsigned)ga;                         // global_addr[31:0]
    g0[3] = (unsigned)((ga >> 32) & 0x01FFFFFFu) | (2u << 30);  // addr[56:32] | type=2
    v8i g1;
    g1[0] = (int)(2u << 16);                      // wg_mask=0, data_size=2 (4 bytes)
    g1[1] = (int)((nelem & 0xFFFFu) << 16);       // tensor_dim0[15:0] in [31:16]
    g1[2] = (int)((nelem >> 16) & 0xFFFFu) | (int)(1u << 16); // td0[31:16] | td1=1
    g1[3] = (int)((nelem & 0xFFFFu) << 16);       // td1[31:16]=0 | tile_dim0=nelem
    g1[4] = 0;                                    // tile_dim1=0(unused), tile_dim2=0
    g1[5] = (int)nelem;                           // tensor_dim0_stride[31:0]
    g1[6] = (int)((nelem & 0xFFFFu) << 16);       // t0s[47:32]=0 | t1s[15:0]
    g1[7] = 0;                                    // t1s[47:16]
    asm volatile("tensor_load_to_lds %0, %1" :: "s"(g0), "s"(g1) : "memory");
}

__global__ __launch_bounds__(256, 1)
void fa_l2_sink_kernel(const float* __restrict__ q,
                       const float* __restrict__ k,
                       const float* __restrict__ v,
                       float* __restrict__ out)
{
    // f32 DMA stage: K tile then V tile (32 KB)
    __shared__ __align__(16) float    lds_stage[2 * TILE_ELEMS];
    __shared__ __align__(16) _Float16 lds_k [BLOCK_N * DHEAD];   // [key][d]   8 KB
    __shared__ __align__(16) _Float16 lds_vt[DHEAD * BLOCK_N];   // [d][key]   8 KB
    __shared__ __align__(16) _Float16 lds_p [NWAVES * 16 * 32];  // P slice    8 KB
    __shared__ __align__(16) float    lds_ksq[BLOCK_N];

    const int tid  = threadIdx.x;
    const int lane = tid & 31;
    const int wave = tid >> 5;
    const int col  = lane & 15;
    const int hi   = (lane >> 4) & 1;

    // wave-uniform scalar copies -> SALU branches (no EXEC masking around WMMA)
    const int wave_u = __builtin_amdgcn_readfirstlane(wave);

    const int  mt   = blockIdx.x & 15;          // 16 m-tiles per head
    const int  bh   = blockIdx.x >> 4;          // B*H = 32
    const long base = (long)bh * N_SEQ * DHEAD;

    const int row0w     = mt * BLOCK_M + wave * 16; // wave's first query row
    const int row_max_u = __builtin_amdgcn_readfirstlane(row0w + 15);

    const unsigned stage_k_off = (unsigned)(unsigned long long)(void*)&lds_stage[0];
    const unsigned stage_v_off = (unsigned)(unsigned long long)(void*)&lds_stage[TILE_ELEMS];

    // ---- Q fragments (A layout), prescaled by 2*SC; |q|^2 kept in f32 ----
    v16h qf[2];
    float qsq = 0.f;
    {
        const float* qrow = q + base + (long)(row0w + col) * DHEAD;
#pragma unroll
        for (int s = 0; s < 2; ++s) {
#pragma unroll
            for (int h2 = 0; h2 < 8; ++h2) {
                int d0 = s * 32 + 2 * (h2 & 3) + ((h2 >= 4) ? 16 : 0) + 8 * hi;
                float2 qq = *(const float2*)(qrow + d0);
                qsq += qq.x * qq.x + qq.y * qq.y;
                qf[s][2 * h2 + 0] = (_Float16)(qq.x * (2.f * SC));
                qf[s][2 * h2 + 1] = (_Float16)(qq.y * (2.f * SC));
            }
        }
        qsq += __shfl_xor(qsq, 16, 32);   // lane holds |q|^2 of row (lane%16)
    }

    // per-lane softmax state for rows r_e = e + 8*hi (matches C/D fragment rows)
    float qsq_r[8], m_i[8], l_i[8];
#pragma unroll
    for (int e = 0; e < 8; ++e) {
        qsq_r[e] = __shfl(qsq, e + 8 * hi, 32);
        m_i[e]   = -SC * qsq_r[e];   // zero-KV sink score
        l_i[e]   = 1.0f;             // sink contribution at m = s_sink
    }
    v8f o_acc[4] = {};               // 16 x 64 f32 output accumulator

    const int num_kt = 2 * mt + 2;

    const int skey = tid >> 2;           // convert pass: key 0..63
    const int sd0  = (tid & 3) * 16;     // convert pass: d base

    // prologue: DMA tile 0 (wave0 -> K, wave1 -> V; TENSORcnt per wave)
    if (wave_u == 0)      tdm_load_1d(k + base, stage_k_off, TILE_ELEMS);
    else if (wave_u == 1) tdm_load_1d(v + base, stage_v_off, TILE_ELEMS);

    for (int kt = 0; kt < num_kt; ++kt) {
        __builtin_amdgcn_s_wait_tensorcnt((short)0);   // no-op for waves 2..7
        __syncthreads();                 // stage(kt) visible; prev compute done

        // ---- convert pass: f32 stage -> f16 tiles + ksq ----
        float ksq_p = 0.f;
#pragma unroll
        for (int i = 0; i < 4; ++i) {
            int dd = sd0 + i * 4;
            float4 kr = *(const float4*)&lds_stage[skey * DHEAD + dd];
            float4 vr = *(const float4*)&lds_stage[TILE_ELEMS + skey * DHEAD + dd];
            ksq_p += kr.x * kr.x + kr.y * kr.y + kr.z * kr.z + kr.w * kr.w;
            *(unsigned*)&lds_k[skey * DHEAD + dd]     = pack2h((_Float16)kr.x, (_Float16)kr.y);
            *(unsigned*)&lds_k[skey * DHEAD + dd + 2] = pack2h((_Float16)kr.z, (_Float16)kr.w);
            lds_vt[(dd + 0) * BLOCK_N + skey] = (_Float16)vr.x;
            lds_vt[(dd + 1) * BLOCK_N + skey] = (_Float16)vr.y;
            lds_vt[(dd + 2) * BLOCK_N + skey] = (_Float16)vr.z;
            lds_vt[(dd + 3) * BLOCK_N + skey] = (_Float16)vr.w;
        }
        ksq_p += __shfl_xor(ksq_p, 1, 32);
        ksq_p += __shfl_xor(ksq_p, 2, 32);
        if ((tid & 3) == 0) lds_ksq[skey] = ksq_p;

        __syncthreads();                 // f16 tiles ready; stage free for reuse

        // pipeline: DMA tile kt+1 into stage while we compute tile kt
        if (kt + 1 < num_kt) {
            const long nb = base + (long)((kt + 1) * BLOCK_N) * DHEAD;
            if (wave_u == 0)      tdm_load_1d(k + nb, stage_k_off, TILE_ELEMS);
            else if (wave_u == 1) tdm_load_1d(v + nb, stage_v_off, TILE_ELEMS);
        }

        if (kt * BLOCK_N <= row_max_u) { // scalar causal skip (wave-uniform)
            // ---- S = (2*SC*Q) @ K^T via WMMA f16 -> f32 ----
            v8f s_acc[4];
#pragma unroll
            for (int n = 0; n < 4; ++n) {
                v8f acc = {};
                int key = n * 16 + col;
#pragma unroll
                for (int s = 0; s < 2; ++s) {
                    v16h bf;
#pragma unroll
                    for (int h2 = 0; h2 < 8; ++h2) {
                        int d0 = s * 32 + 2 * (h2 & 3) + ((h2 >= 4) ? 16 : 0) + 8 * hi;
                        ((unsigned*)&bf)[h2] = *(const unsigned*)&lds_k[key * DHEAD + d0];
                    }
                    acc = __builtin_amdgcn_wmma_f32_16x16x32_f16(
                              false, qf[s], false, bf, (short)0, acc, false, false);
                }
                s_acc[n] = acc;
            }

            float ksq_c[4];
#pragma unroll
            for (int n = 0; n < 4; ++n) ksq_c[n] = lds_ksq[n * 16 + col];

            // ---- bias + causal mask + online softmax (row = e + 8*hi) ----
            float alpha[8];
#pragma unroll
            for (int e = 0; e < 8; ++e) {
                int rowg = row0w + e + 8 * hi;
                float mx = NEGBIG;
                float sv[4];
#pragma unroll
                for (int n = 0; n < 4; ++n) {
                    float x = s_acc[n][e] - SC * (qsq_r[e] + ksq_c[n]);
                    int keyg = kt * BLOCK_N + n * 16 + col;
                    if (keyg > rowg) x = NEGBIG;
                    sv[n] = x;
                    mx = fmaxf(mx, x);
                }
                mx = fmaxf(mx, __shfl_xor(mx, 1, 32));
                mx = fmaxf(mx, __shfl_xor(mx, 2, 32));
                mx = fmaxf(mx, __shfl_xor(mx, 4, 32));
                mx = fmaxf(mx, __shfl_xor(mx, 8, 32));
                float m_new = fmaxf(m_i[e], mx);
                float a = __builtin_exp2f((m_i[e] - m_new) * LOG2E);
                float rs = 0.f;
#pragma unroll
                for (int n = 0; n < 4; ++n) {
                    float p = __builtin_exp2f((sv[n] - m_new) * LOG2E);
                    s_acc[n][e] = p;
                    rs += p;
                }
                rs += __shfl_xor(rs, 1, 32);
                rs += __shfl_xor(rs, 2, 32);
                rs += __shfl_xor(rs, 4, 32);
                rs += __shfl_xor(rs, 8, 32);
                l_i[e]  = l_i[e] * a + rs;
                m_i[e]  = m_new;
                alpha[e] = a;
            }

            // rescale O before accumulating this tile
#pragma unroll
            for (int n = 0; n < 4; ++n)
#pragma unroll
                for (int e = 0; e < 8; ++e)
                    o_acc[n][e] *= alpha[e];

            // ---- O += P @ V, one 32-key slice at a time (small P buffer) ----
            _Float16* pwave = &lds_p[wave * 16 * 32];
#pragma unroll
            for (int s = 0; s < 2; ++s) {
                // C-layout P slice -> per-wave LDS (rows 16 x keys 32)
#pragma unroll
                for (int e = 0; e < 8; ++e) {
                    int prow = e + 8 * hi;
#pragma unroll
                    for (int nn = 0; nn < 2; ++nn)
                        pwave[prow * 32 + nn * 16 + col] =
                            (_Float16)s_acc[2 * s + nn][e];
                }
                // A-layout fragment of P slice (DS in-order within wave)
                v16h pf;
#pragma unroll
                for (int h2 = 0; h2 < 8; ++h2) {
                    int kk = 2 * (h2 & 3) + ((h2 >= 4) ? 16 : 0) + 8 * hi;
                    ((unsigned*)&pf)[h2] = *(const unsigned*)&pwave[col * 32 + kk];
                }
#pragma unroll
                for (int dn = 0; dn < 4; ++dn) {
                    int d = dn * 16 + col;
                    v16h vf;
#pragma unroll
                    for (int h2 = 0; h2 < 8; ++h2) {
                        int keyk = s * 32 + 2 * (h2 & 3) + ((h2 >= 4) ? 16 : 0) + 8 * hi;
                        ((unsigned*)&vf)[h2] = *(const unsigned*)&lds_vt[d * BLOCK_N + keyk];
                    }
                    o_acc[dn] = __builtin_amdgcn_wmma_f32_16x16x32_f16(
                                    false, pf, false, vf, (short)0, o_acc[dn], false, false);
                }
            }
        }
    }

    // ---- epilogue: divide by softmax sum, store f32 ----
    float inv[8];
#pragma unroll
    for (int e = 0; e < 8; ++e) inv[e] = 1.0f / l_i[e];
#pragma unroll
    for (int e = 0; e < 8; ++e) {
        long rb = base + (long)(row0w + e + 8 * hi) * DHEAD;
#pragma unroll
        for (int dn = 0; dn < 4; ++dn)
            out[rb + dn * 16 + col] = o_acc[dn][e] * inv[e];
    }
}

extern "C" void kernel_launch(void* const* d_in, const int* in_sizes, int n_in,
                              void* d_out, int out_size, void* d_ws, size_t ws_size,
                              hipStream_t stream) {
    const float* q = (const float*)d_in[0];
    const float* k = (const float*)d_in[1];
    const float* v = (const float*)d_in[2];
    float* out = (float*)d_out;
    // grid: 16 m-tiles x (B*H = 32) heads = 512 blocks, 256 threads (8 wave32)
    fa_l2_sink_kernel<<<dim3(512), dim3(256), 0, stream>>>(q, k, v, out);
}
